// LaneATT_40699110096907
// MI455X (gfx1250) — matrix operand order repaired and checked by reading, not compiled, not run on hardware
//
#include <hip/hip_runtime.h>
#include <hip/hip_bf16.h>
#include <cstdint>

// ---------------- problem constants ----------------
constexpr int B_   = 4;
constexpr int CIN  = 512;
constexpr int C_   = 64;
constexpr int H_   = 11;
constexpr int W_   = 20;
constexpr int P    = 2784;          // proposals (multiple of 32)
constexpr int S_   = 72;
constexpr int D    = C_ * H_;       // 704 = 22*32 = 11*64
constexpr int PP   = 2784;          // scores padded width (>= P-1, mult of 16)
constexpr int PM1  = P - 1;         // 2783 valid attention columns
constexpr int ACOLS = 5 + S_;       // 77
constexpr int PPAD = 2816;          // row padding for guard-free tiles (mult of 128)

// GEMM tiling
constexpr int TK = 32;
constexpr int BSTRIDE = TK + 8;     // LDS [n][k] row stride (elems); 80B keeps 16B align

typedef __attribute__((ext_vector_type(8)))  float  v8f;
typedef __attribute__((ext_vector_type(16))) __bf16 v16bf;

union FragAB { uint4 u[2]; v16bf v; };

__device__ __forceinline__ void async_load_b128(uint32_t lds_off, const void* gptr) {
    uint64_t ga = (uint64_t)(uintptr_t)gptr;
    asm volatile("global_load_async_to_lds_b128 %0, %1, off"
                 :: "v"(lds_off), "v"(ga) : "memory");
}
template<int N>
__device__ __forceinline__ void wait_async() {
    asm volatile("s_wait_asynccnt %0" :: "n"(N) : "memory");
}
__device__ __forceinline__ uint32_t lds_off32(const void* p) {
    return (uint32_t)(uintptr_t)p;   // flat LDS aperture: low 32 bits = LDS offset
}

// ---------------- 1x1 conv ----------------
__global__ __launch_bounds__(256) void conv1x1_kernel(
    const float* __restrict__ feat, const float* __restrict__ w,
    const float* __restrict__ bias, float* __restrict__ f)
{
    int idx = blockIdx.x * blockDim.x + threadIdx.x;
    if (idx >= B_ * C_ * H_ * W_) return;
    int ww = idx % W_; int tmp = idx / W_;
    int h  = tmp % H_; tmp /= H_;
    int c  = tmp % C_; int b = tmp / C_;
    float s = bias[c];
    const float* fp = feat + ((size_t)b * CIN) * (H_ * W_) + h * W_ + ww;
    const float* wp = w + (size_t)c * CIN;
    #pragma unroll 4
    for (int i = 0; i < CIN; ++i) s += fp[(size_t)i * (H_ * W_)] * wp[i];
    f[idx] = s;
}

// ------- gather rois -> baf32 [B][P][D], baf16 [B][PPAD][D], bafT16 [B][D][P] -------
__global__ __launch_bounds__(256) void build_baf_kernel(
    const float* __restrict__ f, const int* __restrict__ cut_xs,
    float* __restrict__ baf32, __bf16* __restrict__ baf16, __bf16* __restrict__ bafT16)
{
    size_t idx = (size_t)blockIdx.x * blockDim.x + threadIdx.x;
    if (idx >= (size_t)B_ * PPAD * D) return;
    int d = idx % D; size_t tmp = idx / D;
    int p = tmp % PPAD; int b = tmp / PPAD;
    if (p >= P) { baf16[idx] = (__bf16)0.0f; return; }   // zero pad rows
    int c = d / H_, h = d % H_;
    int cx = cut_xs[p * H_ + h];
    float v = 0.0f;
    if (cx >= 0 && cx <= W_) {
        int x = cx < W_ ? cx : (W_ - 1);
        v = f[(((size_t)b * C_ + c) * H_ + h) * W_ + x];
    }
    baf32[((size_t)b * P + p) * D + d] = v;
    baf16[idx] = (__bf16)v;
    bafT16[((size_t)b * D + d) * P + p] = (__bf16)v;
}

// ---------------- att_w -> bf16 padded to PPAD rows ----------------
__global__ __launch_bounds__(256) void attw_bf16_kernel(
    const float* __restrict__ att_w, __bf16* __restrict__ attw16)
{
    size_t idx = (size_t)blockIdx.x * blockDim.x + threadIdx.x;
    if (idx >= (size_t)PPAD * D) return;
    int n = idx / D;
    attw16[idx] = (n < PM1) ? (__bf16)att_w[idx] : (__bf16)0.0f;
}

// ------- GEMM1: scores[b] (PxPP) = baf16[b] (PPADxD) @ attw16^T + att_b -------
// 64(M) x 128(N) x 32(K) tile; 128 threads = 4 waves; double-buffered async staging
__global__ __launch_bounds__(128) void gemm_scores_kernel(
    const __bf16* __restrict__ baf, const __bf16* __restrict__ attw,
    const float* __restrict__ att_b, float* __restrict__ scores)
{
    __shared__ __align__(16) __bf16 sA[2][64 * TK];         // [m][k]
    __shared__ __align__(16) __bf16 sB[2][128 * BSTRIDE];   // [n][k]

    const int b  = blockIdx.z;
    const int M0 = blockIdx.x * 64;
    const int N0 = blockIdx.y * 128;
    const int t  = threadIdx.x;
    const int lane = t & 31, wave = t >> 5;
    const int half = lane >> 4, ln = lane & 15;

    v8f acc[8] = {};

    const __bf16* Ab = baf  + ((size_t)b * PPAD + M0) * D;
    const __bf16* Bb = attw + (size_t)N0 * D;

    const int am0 = t >> 2, aseg = t & 3, am1 = am0 + 32;
    const uint32_t sAoff[2] = { lds_off32(&sA[0][0]), lds_off32(&sA[1][0]) };
    const uint32_t sBoff[2] = { lds_off32(&sB[0][0]), lds_off32(&sB[1][0]) };

    auto issue = [&](int k0, int pb) {
        async_load_b128(sAoff[pb] + (uint32_t)((am0 * TK + aseg * 8) * 2),
                        Ab + (size_t)am0 * D + k0 + aseg * 8);
        async_load_b128(sAoff[pb] + (uint32_t)((am1 * TK + aseg * 8) * 2),
                        Ab + (size_t)am1 * D + k0 + aseg * 8);
        #pragma unroll
        for (int q = 0; q < 4; ++q) {
            int c = t + q * 128, n = c >> 2, seg = c & 3;
            async_load_b128(sBoff[pb] + (uint32_t)((n * BSTRIDE + seg * 8) * 2),
                            Bb + (size_t)n * D + k0 + seg * 8);
        }
    };

    issue(0, 0);
    int pb = 0;
    for (int k0 = 0; k0 < D; k0 += TK) {
        if (k0 + TK < D) { issue(k0 + TK, pb ^ 1); wait_async<6>(); }
        else             { wait_async<0>(); }
        __syncthreads();                       // tile pb fully in LDS for all waves

        const __bf16* cA = &sA[pb][0];
        const __bf16* cB = &sB[pb][0];
        FragAB fa;
        const int am = wave * 16 + ln;
        fa.u[0] = *reinterpret_cast<const uint4*>(&cA[am * TK + half * 16]);
        fa.u[1] = *reinterpret_cast<const uint4*>(&cA[am * TK + half * 16 + 8]);
        #pragma unroll
        for (int g = 0; g < 2; ++g) {
            FragAB fb[4];
            #pragma unroll
            for (int nt = 0; nt < 4; ++nt) {
                const int bn = (g * 4 + nt) * 16 + ln;
                fb[nt].u[0] = *reinterpret_cast<const uint4*>(&cB[bn * BSTRIDE + half * 16]);
                fb[nt].u[1] = *reinterpret_cast<const uint4*>(&cB[bn * BSTRIDE + half * 16 + 8]);
            }
            #pragma unroll
            for (int nt = 0; nt < 4; ++nt)
                acc[g * 4 + nt] = __builtin_amdgcn_wmma_f32_16x16x32_bf16(
                    false, fa.v, false, fb[nt].v, (short)0, acc[g * 4 + nt], false, false);
        }
        __syncthreads();                       // all reads of tile pb done before overwrite
        pb ^= 1;
    }

    #pragma unroll
    for (int j = 0; j < 8; ++j) {
        int gc = N0 + j * 16 + ln;
        if (gc >= PP) continue;
        float bias = (gc < PM1) ? att_b[gc] : 0.0f;
        #pragma unroll
        for (int r = 0; r < 8; ++r) {
            int gr = M0 + wave * 16 + half * 8 + r;
            if (gr < P)
                scores[((size_t)b * P + gr) * PP + gc] = acc[j][r] + bias;
        }
    }
}

// ---------------- softmax over PM1 cols, in place ----------------
__global__ __launch_bounds__(256) void softmax_kernel(float* __restrict__ scores)
{
    float* rp = scores + (size_t)blockIdx.x * PP;
    __shared__ float red[256];
    const int t = threadIdx.x;

    float m = -3.4e38f;
    for (int j = t; j < PM1; j += 256) m = fmaxf(m, rp[j]);
    red[t] = m; __syncthreads();
    for (int s = 128; s > 0; s >>= 1) { if (t < s) red[t] = fmaxf(red[t], red[t + s]); __syncthreads(); }
    m = red[0]; __syncthreads();

    float sum = 0.0f;
    for (int j = t; j < PM1; j += 256) { float e = __expf(rp[j] - m); rp[j] = e; sum += e; }
    red[t] = sum; __syncthreads();
    for (int s = 128; s > 0; s >>= 1) { if (t < s) red[t] += red[t + s]; __syncthreads(); }
    float inv = 1.0f / red[0];
    for (int j = t; j < PM1; j += 256) rp[j] *= inv;
}

// ------- build M (bf16, padded rows): M[i][j] = (j==i) ? 0 : att[i][j - (j>i)] -------
__global__ __launch_bounds__(256) void build_M_kernel(
    const float* __restrict__ att, __bf16* __restrict__ M16)
{
    size_t idx = (size_t)blockIdx.x * blockDim.x + threadIdx.x;
    if (idx >= (size_t)B_ * PPAD * P) return;
    int j = idx % P; size_t tmp = idx / P;
    int i = tmp % PPAD; int b = tmp / PPAD;
    float v = 0.0f;
    if (i < P && j != i) {
        int col = j - (j > i ? 1 : 0);        // <= P-2
        v = att[((size_t)b * P + i) * PP + col];
    }
    M16[idx] = (__bf16)v;
}

// ------- GEMM2: att_feats[b] (PxD) = M16[b] (PPADxP) @ baf (PxD), B via bafT -------
// 64(M) x 64(N) x 32(K) tile; 128 threads = 4 waves; double-buffered async staging
__global__ __launch_bounds__(128) void gemm_attfeats_kernel(
    const __bf16* __restrict__ M16, const __bf16* __restrict__ bafT,
    float* __restrict__ attf)
{
    __shared__ __align__(16) __bf16 sA[2][64 * TK];       // [m][k]
    __shared__ __align__(16) __bf16 sB[2][64 * BSTRIDE];  // [n][k]

    const int b  = blockIdx.z;
    const int M0 = blockIdx.x * 64;
    const int N0 = blockIdx.y * 64;          // D = 11*64 exact
    const int t  = threadIdx.x;
    const int lane = t & 31, wave = t >> 5;
    const int half = lane >> 4, ln = lane & 15;

    v8f acc[4] = {};

    const __bf16* Ab  = M16  + ((size_t)b * PPAD + M0) * P;
    const __bf16* BTb = bafT + ((size_t)b * D + N0) * P;

    const int r0 = t >> 2, seg = t & 3, r1 = r0 + 32;
    const uint32_t sAoff[2] = { lds_off32(&sA[0][0]), lds_off32(&sA[1][0]) };
    const uint32_t sBoff[2] = { lds_off32(&sB[0][0]), lds_off32(&sB[1][0]) };

    auto issue = [&](int k0, int pb) {
        async_load_b128(sAoff[pb] + (uint32_t)((r0 * TK + seg * 8) * 2),
                        Ab + (size_t)r0 * P + k0 + seg * 8);
        async_load_b128(sAoff[pb] + (uint32_t)((r1 * TK + seg * 8) * 2),
                        Ab + (size_t)r1 * P + k0 + seg * 8);
        async_load_b128(sBoff[pb] + (uint32_t)((r0 * BSTRIDE + seg * 8) * 2),
                        BTb + (size_t)r0 * P + k0 + seg * 8);
        async_load_b128(sBoff[pb] + (uint32_t)((r1 * BSTRIDE + seg * 8) * 2),
                        BTb + (size_t)r1 * P + k0 + seg * 8);
    };

    issue(0, 0);
    int pb = 0;
    for (int k0 = 0; k0 < P; k0 += TK) {     // K = P = 87*32 exact
        if (k0 + TK < P) { issue(k0 + TK, pb ^ 1); wait_async<4>(); }
        else             { wait_async<0>(); }
        __syncthreads();

        const __bf16* cA = &sA[pb][0];
        const __bf16* cB = &sB[pb][0];
        FragAB fa;
        const int am = wave * 16 + ln;
        fa.u[0] = *reinterpret_cast<const uint4*>(&cA[am * TK + half * 16]);
        fa.u[1] = *reinterpret_cast<const uint4*>(&cA[am * TK + half * 16 + 8]);
        FragAB fb[4];
        #pragma unroll
        for (int nt = 0; nt < 4; ++nt) {
            const int bn = nt * 16 + ln;
            fb[nt].u[0] = *reinterpret_cast<const uint4*>(&cB[bn * BSTRIDE + half * 16]);
            fb[nt].u[1] = *reinterpret_cast<const uint4*>(&cB[bn * BSTRIDE + half * 16 + 8]);
        }
        #pragma unroll
        for (int nt = 0; nt < 4; ++nt)
            acc[nt] = __builtin_amdgcn_wmma_f32_16x16x32_bf16(
                false, fa.v, false, fb[nt].v, (short)0, acc[nt], false, false);
        __syncthreads();
        pb ^= 1;
    }

    #pragma unroll
    for (int nt = 0; nt < 4; ++nt) {
        int gc = N0 + nt * 16 + ln;
        #pragma unroll
        for (int r = 0; r < 8; ++r) {
            int gr = M0 + wave * 16 + half * 8 + r;
            if (gr < P)
                attf[((size_t)b * P + gr) * D + gc] = acc[nt][r];
        }
    }
}

// ---------------- head: feats = [attf | baf], projections + assemble ----------------
__global__ __launch_bounds__(128) void head_kernel(
    const float* __restrict__ attf, const float* __restrict__ baf,
    const float* __restrict__ cls_w, const float* __restrict__ cls_b,
    const float* __restrict__ reg_w, const float* __restrict__ reg_b,
    const float* __restrict__ anchors, float* __restrict__ out)
{
    __shared__ float row[2 * D];
    const int bp = blockIdx.x;               // b*P + p
    const int p  = bp % P;
    const int t  = threadIdx.x;

    const float* a1 = attf + (size_t)bp * D;
    const float* a2 = baf  + (size_t)bp * D;
    for (int j = t; j < D; j += 128) { row[j] = a1[j]; row[D + j] = a2[j]; }
    __syncthreads();

    float* orow = out + (size_t)bp * ACOLS;
    const float* arow = anchors + (size_t)p * ACOLS;

    if (t < 75) {
        const float* wrow; float bias;
        if (t < 2) { wrow = cls_w + (size_t)t * (2 * D);       bias = cls_b[t]; }
        else       { wrow = reg_w + (size_t)(t - 2) * (2 * D); bias = reg_b[t - 2]; }
        float acc = bias;
        #pragma unroll 4
        for (int j = 0; j < 2 * D; ++j) acc += row[j] * wrow[j];
        if (t < 2) orow[t]     = acc;                  // cols 0,1
        else       orow[t + 2] = arow[t + 2] + acc;    // cols 4..76
    } else if (t < 77) {
        orow[t - 73] = arow[t - 73];                   // cols 2,3 passthrough
    }
}

// ---------------- host side ----------------
static inline size_t alignUp(size_t x, size_t a) { return (x + a - 1) & ~(a - 1); }

extern "C" void kernel_launch(void* const* d_in, const int* in_sizes, int n_in,
                              void* d_out, int out_size, void* d_ws, size_t ws_size,
                              hipStream_t stream) {
    (void)in_sizes; (void)n_in; (void)out_size; (void)ws_size;

    const float* feat    = (const float*)d_in[0];
    const int*   cut_xs  = (const int*)  d_in[1];
    const float* conv1_w = (const float*)d_in[2];
    const float* conv1_b = (const float*)d_in[3];
    const float* att_w   = (const float*)d_in[4];
    const float* att_b   = (const float*)d_in[5];
    const float* cls_w   = (const float*)d_in[6];
    const float* cls_b   = (const float*)d_in[7];
    const float* reg_w   = (const float*)d_in[8];
    const float* reg_b   = (const float*)d_in[9];
    const float* anchors = (const float*)d_in[10];
    float* out = (float*)d_out;

    char* ws = (char*)d_ws;
    size_t off = 0;
    auto carve = [&](size_t bytes) -> void* {
        void* pp = ws + off;
        off = alignUp(off + bytes, 256);
        return pp;
    };

    float*  f_buf  = (float*) carve((size_t)B_ * C_ * H_ * W_ * sizeof(float));
    float*  baf32  = (float*) carve((size_t)B_ * P * D * sizeof(float));
    __bf16* baf16  = (__bf16*)carve((size_t)B_ * PPAD * D * sizeof(__bf16));
    __bf16* bafT16 = (__bf16*)carve((size_t)B_ * D * P * sizeof(__bf16));
    __bf16* attw16 = (__bf16*)carve((size_t)PPAD * D * sizeof(__bf16));
    float*  scores = (float*) carve((size_t)B_ * P * PP * sizeof(float));
    __bf16* M16    = (__bf16*)carve((size_t)B_ * PPAD * P * sizeof(__bf16));
    float*  attf   = (float*) carve((size_t)B_ * P * D * sizeof(float));

    // 1) 1x1 conv
    {
        int total = B_ * C_ * H_ * W_;
        conv1x1_kernel<<<(total + 255) / 256, 256, 0, stream>>>(feat, conv1_w, conv1_b, f_buf);
    }
    // 2) gather -> baf (fp32, bf16 padded, bf16 transposed)
    {
        size_t total = (size_t)B_ * PPAD * D;
        build_baf_kernel<<<(unsigned)((total + 255) / 256), 256, 0, stream>>>(
            f_buf, cut_xs, baf32, baf16, bafT16);
    }
    // 3) att_w -> bf16 (padded)
    {
        size_t total = (size_t)PPAD * D;
        attw_bf16_kernel<<<(unsigned)((total + 255) / 256), 256, 0, stream>>>(att_w, attw16);
    }
    // 4) scores GEMM (WMMA bf16, double-buffered async staging)
    {
        dim3 grid((P + 63) / 64, PPAD / 128, B_);
        gemm_scores_kernel<<<grid, 128, 0, stream>>>(baf16, attw16, att_b, scores);
    }
    // 5) softmax in place
    softmax_kernel<<<B_ * P, 256, 0, stream>>>(scores);
    // 6) M matrix (bf16, shifted/diag-zero attention)
    {
        size_t total = (size_t)B_ * PPAD * P;
        build_M_kernel<<<(unsigned)((total + 255) / 256), 256, 0, stream>>>(scores, M16);
    }
    // 7) att_feats GEMM (WMMA bf16, double-buffered async staging)
    {
        dim3 grid((P + 63) / 64, D / 64, B_);
        gemm_attfeats_kernel<<<grid, 128, 0, stream>>>(M16, bafT16, attf);
    }
    // 8) head + assemble proposals
    head_kernel<<<B_ * P, 128, 0, stream>>>(attf, baf32, cls_w, cls_b, reg_w, reg_b, anchors, out);
}
